// BahdanauAttention1D_90563680403619
// MI455X (gfx1250) — compile-verified, matrix operand order checked by reading, not run on hardware
//
#include <hip/hip_runtime.h>
#include <hip/hip_bf16.h>

#define B_ 64
#define T_ 8192
#define D_ 256
#define U_ 256

typedef __attribute__((ext_vector_type(16))) __bf16 v16bf;
typedef __attribute__((ext_vector_type(8)))  float  v8f;

#if __has_builtin(__builtin_amdgcn_tanhf)
#define FAST_TANH(x) __builtin_amdgcn_tanhf(x)
#elif __has_builtin(__builtin_amdgcn_tanh_f32)
#define FAST_TANH(x) __builtin_amdgcn_tanh_f32(x)
#else
#define FAST_TANH(x) tanhf(x)
#endif

__device__ __forceinline__ unsigned short f32_bf16_rne(float f) {
  unsigned u = __builtin_bit_cast(unsigned, f);
  unsigned r = u + 0x7FFFu + ((u >> 16) & 1u);
  return (unsigned short)(r >> 16);
}

// Pack two f32 -> one dword holding (lo=bf16(x), hi=bf16(y)), RNE
__device__ __forceinline__ unsigned pack_bf16(float x, float y) {
  return (unsigned)f32_bf16_rne(x) | ((unsigned)f32_bf16_rne(y) << 16);
}

union AFrag {
  v16bf v;
  unsigned w[8];
};

// K index within a 32-wide K-slice for WMMA 16-bit A/B fragment element e on `lane`
__device__ __forceinline__ int kmap(int lane, int e) {
  int base = (e < 8) ? ((lane < 16) ? 0 : 8) : ((lane < 16) ? 16 : 24);
  return base + (e & 7);
}

__device__ __forceinline__ AFrag load_a_frag(const float* arow, int b1, int b2) {
  float4 f0 = *(const float4*)(arow + b1);
  float4 f1 = *(const float4*)(arow + b1 + 4);
  float4 f2 = *(const float4*)(arow + b2);
  float4 f3 = *(const float4*)(arow + b2 + 4);
  AFrag a;
  a.w[0] = pack_bf16(f0.x, f0.y);
  a.w[1] = pack_bf16(f0.z, f0.w);
  a.w[2] = pack_bf16(f1.x, f1.y);
  a.w[3] = pack_bf16(f1.z, f1.w);
  a.w[4] = pack_bf16(f2.x, f2.y);
  a.w[5] = pack_bf16(f2.z, f2.w);
  a.w[6] = pack_bf16(f3.x, f3.y);
  a.w[7] = pack_bf16(f3.z, f3.w);
  return a;
}

// ---------------------------------------------------------------------------
// Kernel 0: zero accumulators + pack W (fp32 [D,U]) into bf16 WMMA-B fragments
// Layout: wb[((kb*16 + n)*32 + lane)*16 + e]  (kb: K/32 slice, n: U/16 tile)
// ---------------------------------------------------------------------------
__global__ void __launch_bounds__(256)
prep_kernel(const float* __restrict__ W, unsigned short* __restrict__ wb,
            float* __restrict__ sum_e, float* __restrict__ accum) {
  int id = blockIdx.x * blockDim.x + threadIdx.x;   // 65536 threads
  if (id < B_)        sum_e[id] = 0.0f;
  if (id < B_ * U_)   accum[id] = 0.0f;
  int e    = id & 15;
  int lane = (id >> 4) & 31;
  int n    = (id >> 9) & 15;
  int kb   = id >> 13;
  int K = kb * 32 + kmap(lane, e);
  int u = n * 16 + (lane & 15);
  wb[id] = f32_bf16_rne(W[K * U_ + u]);
}

// ---------------------------------------------------------------------------
// Kernel 1: fused GEMM + tanh-score + exp + online weighted accumulation.
// Block: 256 threads (8 waves). Packed W staged into LDS once per block.
// Each wave owns TWO 16-row t-tiles (32 rows); every B-fragment read from LDS
// is reused by two v_wmma_f32_16x16x32_bf16 -> halves LDS bandwidth demand.
// ---------------------------------------------------------------------------
__global__ void __launch_bounds__(256)
attn_main(const float* __restrict__ X, const unsigned char* __restrict__ msk,
          const unsigned short* __restrict__ wb,
          const float* __restrict__ bias, const float* __restrict__ vvec,
          float* __restrict__ e_out, float* __restrict__ sum_e,
          float* __restrict__ accum) {
  __shared__ float s_acc[U_];
  __shared__ v16bf s_wb[4096];          // 128 KiB packed W (8 kb x 16 n x 32 lanes)

  const int tid = threadIdx.x;
  s_acc[tid] = 0.0f;
  {
    const v16bf* g = (const v16bf*)wb;
#pragma unroll
    for (int i = 0; i < 16; ++i)
      s_wb[i * 256 + tid] = g[i * 256 + tid];
  }
  __syncthreads();

  const int lane = tid & 31;
  const int wave = tid >> 5;
  const int bb   = blockIdx.y;
  const int col  = lane & 15;      // N (and A-row) within tile
  const int hi   = lane >> 4;      // lane half

  const int t0 = blockIdx.x * 256 + wave * 32;   // this wave's 32 rows
  const float* arow0 = X + ((size_t)bb * T_ + (t0 + col)) * D_;
  const float* arow1 = arow0 + 16 * D_;

  v8f acc[2][16];
#pragma unroll
  for (int t = 0; t < 2; ++t)
#pragma unroll
    for (int n = 0; n < 16; ++n)
      acc[t][n] = v8f{0.f,0.f,0.f,0.f,0.f,0.f,0.f,0.f};

  for (int kb = 0; kb < 8; ++kb) {
    const int kb32 = kb * 32;
    const int b1 = kb32 + (hi ? 8  : 0);
    const int b2 = kb32 + (hi ? 24 : 16);
    if (kb < 7) __builtin_prefetch(arow0 + kb32 + 32, 0, 0);
    AFrag a0 = load_a_frag(arow0, b1, b2);
    AFrag a1 = load_a_frag(arow1, b1, b2);
#pragma unroll
    for (int n = 0; n < 16; ++n) {
      v16bf bm = s_wb[(kb * 16 + n) * 32 + lane];
      acc[0][n] = __builtin_amdgcn_wmma_f32_16x16x32_bf16(
          false, a0.v, false, bm, (short)0, acc[0][n], false, false);
      acc[1][n] = __builtin_amdgcn_wmma_f32_16x16x32_bf16(
          false, a1.v, false, bm, (short)0, acc[1][n], false, false);
    }
  }

  // Per-lane partial scores: sc[t][j] covers row M = j + 8*hi of tile t,
  // summed over this lane's u columns (u = n*16 + col).
  float sc[2][8];
#pragma unroll
  for (int t = 0; t < 2; ++t)
#pragma unroll
    for (int j = 0; j < 8; ++j) sc[t][j] = 0.0f;
#pragma unroll
  for (int n = 0; n < 16; ++n) {
    const int u = n * 16 + col;
    const float bu = bias[u];
    const float vu = vvec[u];
#pragma unroll
    for (int t = 0; t < 2; ++t)
#pragma unroll
      for (int j = 0; j < 8; ++j)
        sc[t][j] += vu * FAST_TANH(bu + acc[t][n][j]);
  }
  // Reduce across the 16 lanes of each half (xor masks stay within halves).
#pragma unroll
  for (int t = 0; t < 2; ++t)
#pragma unroll
    for (int j = 0; j < 8; ++j) {
      float v = sc[t][j];
      v += __shfl_xor(v, 1);
      v += __shfl_xor(v, 2);
      v += __shfl_xor(v, 4);
      v += __shfl_xor(v, 8);
      sc[t][j] = v;
    }

  float ex[2][8];
#pragma unroll
  for (int t = 0; t < 2; ++t) {
    const int rbase = t0 + t * 16 + hi * 8;
#pragma unroll
    for (int j = 0; j < 8; ++j) {
      unsigned char m = msk[(size_t)bb * T_ + rbase + j];
      ex[t][j] = m ? expf(sc[t][j]) : 0.0f;
    }
    if (col == 0) {
#pragma unroll
      for (int j = 0; j < 8; ++j)
        e_out[(size_t)bb * T_ + rbase + j] = ex[t][j];
    }
  }

  float le = 0.0f;
#pragma unroll
  for (int t = 0; t < 2; ++t)
#pragma unroll
    for (int j = 0; j < 8; ++j) le += ex[t][j];
  le += __shfl_xor(le, 16);               // combine row halves
  if (lane == 0) atomicAdd(&sum_e[bb], le);

  // Weighted accumulation: partial over this lane's 16 rows (both tiles),
  // combine halves, then block-level LDS reduction, one global atomic per u.
#pragma unroll
  for (int n = 0; n < 16; ++n) {
    float p = 0.0f;
#pragma unroll
    for (int t = 0; t < 2; ++t)
#pragma unroll
      for (int j = 0; j < 8; ++j) p += ex[t][j] * acc[t][n][j];
    p += __shfl_xor(p, 16);
    if (!hi) atomicAdd(&s_acc[n * 16 + col], p);
  }
  __syncthreads();
  atomicAdd(&accum[(size_t)bb * U_ + tid], s_acc[tid]);
}

// ---------------------------------------------------------------------------
// Kernel 2: normalize -> d_out = [output (B*U) | weights (B*T)]
// ---------------------------------------------------------------------------
__global__ void __launch_bounds__(256)
finalize_kernel(const float* __restrict__ sum_e, const float* __restrict__ accum,
                const float* __restrict__ e_ws, float* __restrict__ out) {
  int i = blockIdx.x * blockDim.x + threadIdx.x;
  const int n_out = B_ * U_;
  const int n_tot = n_out + B_ * T_;
  if (i >= n_tot) return;
  if (i < n_out) {
    int b = i / U_;
    out[i] = accum[i] / sum_e[b];
  } else {
    int j = i - n_out;
    int b = j / T_;
    out[i] = e_ws[j] / sum_e[b];
  }
}

extern "C" void kernel_launch(void* const* d_in, const int* in_sizes, int n_in,
                              void* d_out, int out_size, void* d_ws, size_t ws_size,
                              hipStream_t stream) {
  const float*         X    = (const float*)d_in[0];
  const unsigned char* msk  = (const unsigned char*)d_in[1];
  const float*         W    = (const float*)d_in[2];
  const float*         bias = (const float*)d_in[3];
  const float*         vv   = (const float*)d_in[4];
  float* out = (float*)d_out;

  char* ws = (char*)d_ws;
  float*          sum_e = (float*)(ws + 0);                // 64 f32
  float*          accum = (float*)(ws + 256);              // 16384 f32
  float*          e_ws  = (float*)(ws + 65792);            // B*T f32
  unsigned short* wb    = (unsigned short*)(ws + 2162944); // 65536 bf16

  prep_kernel<<<65536 / 256, 256, 0, stream>>>(W, wb, sum_e, accum);

  dim3 grid(T_ / 256, B_);
  attn_main<<<grid, 256, 0, stream>>>(X, msk, wb, bias, vv, e_ws, sum_e, accum);

  const int n_tot = B_ * U_ + B_ * T_;
  finalize_kernel<<<(n_tot + 255) / 256, 256, 0, stream>>>(sum_e, accum, e_ws, out);
}
